// MoE_88003879895645
// MI455X (gfx1250) — compile-verified, matrix operand order checked by reading, not run on hardware
//
#include <hip/hip_runtime.h>

// ---------------------------------------------------------------------------
// MoE forward for MI455X (gfx1250, wave32, WMMA).
//   B=4096 rows, D=1024 in-dim, O=1024 out-dim, E=16 experts, top-2 gate.
// bf16 hi/lo split GEMM on v_wmma_f32_16x16x32_bf16 (~fp32 accuracy at 3/8
// the cost of fp32 WMMA). 128x128 block tiles, double-buffered LDS staging
// via GLOBAL_LOAD_ASYNC_TO_LDS_B128 when available (ASYNCcnt-tracked DMA,
// no VGPR round-trip), else global->reg->ds_store. All DS/global offsets are
// immediates; fused bias+relu+gate epilogue with non-temporal stores.
// ---------------------------------------------------------------------------

#define BB 4096
#define DD 1024
#define OO 1024
#define EE 16

typedef __bf16 bf16_t;
typedef __attribute__((ext_vector_type(16))) __bf16 v16bf;
typedef __attribute__((ext_vector_type(8)))  __bf16 v8bf;
typedef __attribute__((ext_vector_type(8)))  float  v8f;

#if __has_builtin(__builtin_amdgcn_global_load_async_to_lds_b128) && \
    __has_builtin(__builtin_amdgcn_s_wait_asynccnt)
#define HAVE_ASYNC_LDS 1
#endif

// ---------------------------------------------------------------------------
// Kernel A: split fp32 -> bf16 hi + bf16 lo (lo = bf16(f - float(hi)))
// ---------------------------------------------------------------------------
__global__ void moe_split_bf16(const float* __restrict__ in,
                               bf16_t* __restrict__ hi,
                               bf16_t* __restrict__ lo,
                               int n) {
    int stride = gridDim.x * blockDim.x;
    for (int i = blockIdx.x * blockDim.x + threadIdx.x; i < n; i += stride) {
        float f  = in[i];
        bf16_t h = (bf16_t)f;
        float r  = f - (float)h;
        hi[i] = h;
        lo[i] = (bf16_t)r;
    }
}

// ---------------------------------------------------------------------------
// Kernel B: router logits, top-2, gate[E,B] (E-major for the GEMM epilogue).
// ---------------------------------------------------------------------------
__global__ void moe_router(const float* __restrict__ x,
                           const float* __restrict__ Wr,
                           const float* __restrict__ br,
                           float* __restrict__ logits_out,   // [B,E]
                           float* __restrict__ idx_out,      // [B,2] (as float)
                           float* __restrict__ gate) {       // [E,B]
    int b = blockIdx.x * blockDim.x + threadIdx.x;
    if (b >= BB) return;

    float acc[EE];
#pragma unroll
    for (int e = 0; e < EE; ++e) acc[e] = 0.f;

    const float4* xr = reinterpret_cast<const float4*>(x + (size_t)b * DD);
    for (int d4 = 0; d4 < DD / 4; ++d4) {
        float4 xv = xr[d4];
#pragma unroll
        for (int e = 0; e < EE; ++e) {
            float4 wv = reinterpret_cast<const float4*>(Wr + (size_t)e * DD)[d4];
            acc[e] += xv.x * wv.x + xv.y * wv.y + xv.z * wv.z + xv.w * wv.w;
        }
    }
#pragma unroll
    for (int e = 0; e < EE; ++e) {
        acc[e] += br[e];
        logits_out[(size_t)b * EE + e] = acc[e];
    }

    float v0 = -__builtin_inff(); int i0 = 0;
#pragma unroll
    for (int e = 0; e < EE; ++e) if (acc[e] > v0) { v0 = acc[e]; i0 = e; }
    float v1 = -__builtin_inff(); int i1 = 0;
#pragma unroll
    for (int e = 0; e < EE; ++e) if (e != i0 && acc[e] > v1) { v1 = acc[e]; i1 = e; }

    idx_out[(size_t)b * 2 + 0] = (float)i0;
    idx_out[(size_t)b * 2 + 1] = (float)i1;

#pragma unroll
    for (int e = 0; e < EE; ++e) {
        float g = (e == i0) ? v0 : ((e == i1) ? v1 : 0.f);
        gate[(size_t)e * BB + b] = g;
    }
}

// ---------------------------------------------------------------------------
// Kernel C: expert GEMM, LDS-staged WMMA.
//
// Stage layout (32 KB), "chunk-plane" form: plane c (0..3) holds K in
// [8c, 8c+8) for 128 rows, 16 B per (row, plane) cell:
//   byte(row, c) = c*2048 + row*16
//   A_hi @ 0 | A_lo @ 8192 | B_hi @ 16384 | B_lo @ 24576
// Fragment (ISA 7.12.2): lane half h reads planes {h, h+2} at its row ->
// two ds_load_b128 from one invariant address + immediate offsets.
// ---------------------------------------------------------------------------
#define STAGE_BYTES 32768

__device__ __forceinline__ v16bf frag_ld(const char* p) {
    v8bf c0 = *reinterpret_cast<const v8bf*>(p);          // plane h
    v8bf c1 = *reinterpret_cast<const v8bf*>(p + 4096);   // plane h+2
    return __builtin_shufflevector(c0, c1,
        0,1,2,3,4,5,6,7,8,9,10,11,12,13,14,15);
}

__device__ __forceinline__ void stage_compute(const char* st, int ra, int rb,
                                              v8f acc[4][2]) {
    v16bf ah[4], al[4], bh[2], bl[2];
#pragma unroll
    for (int i = 0; i < 4; ++i) {
        ah[i] = frag_ld(st + ra + i * 256);
        al[i] = frag_ld(st + ra + i * 256 + 8192);
    }
#pragma unroll
    for (int j = 0; j < 2; ++j) {
        bh[j] = frag_ld(st + rb + j * 256);
        bl[j] = frag_ld(st + rb + j * 256 + 8192);
    }
#pragma unroll
    for (int i = 0; i < 4; ++i)
#pragma unroll
        for (int j = 0; j < 2; ++j) {
            acc[i][j] = __builtin_amdgcn_wmma_f32_16x16x32_bf16(
                false, ah[i], false, bh[j], (short)0, acc[i][j], false, false);
            acc[i][j] = __builtin_amdgcn_wmma_f32_16x16x32_bf16(
                false, ah[i], false, bl[j], (short)0, acc[i][j], false, false);
            acc[i][j] = __builtin_amdgcn_wmma_f32_16x16x32_bf16(
                false, al[i], false, bh[j], (short)0, acc[i][j], false, false);
        }
}

#if defined(HAVE_ASYNC_LDS)
// --- async DMA staging path (GLOBAL_LOAD_ASYNC_TO_LDS_B128, ASYNCcnt) ------
// Builtin signature (from hipcc diagnostic): param0 is
// 'int __attribute__((vector_size(16))) __device__ *' (AS1 int4*),
// param1 the AS3 counterpart, then two int immediates (offset, cpol).
typedef int v4i_t __attribute__((vector_size(16)));
typedef __attribute__((address_space(1))) v4i_t gv4i;
typedef __attribute__((address_space(3))) v4i_t lv4i;

__device__ __forceinline__ void async_cp16(const bf16_t* g, char* l) {
    // Generic LDS pointer's low 32 bits are the AS(3) byte offset
    // (ISA 10.2 aperture mapping), so the truncating cast is exact.
    __builtin_amdgcn_global_load_async_to_lds_b128(
        (gv4i*)(unsigned long long)g,
        (lv4i*)(unsigned int)(unsigned long long)l,
        0, 0);
}

__device__ __forceinline__ void stage_fetch_async(char* st, int so,
        const bf16_t* __restrict__ xh, const bf16_t* __restrict__ xl,
        const bf16_t* __restrict__ whe, const bf16_t* __restrict__ wle,
        int gA, int gB, int k0) {
    async_cp16(xh  + gA + k0,      st + so);
    async_cp16(xh  + gA + k0 + 16, st + so + 4096);
    async_cp16(xl  + gA + k0,      st + so + 8192);
    async_cp16(xl  + gA + k0 + 16, st + so + 12288);
    async_cp16(whe + gB + k0,      st + so + 16384);
    async_cp16(whe + gB + k0 + 16, st + so + 20480);
    async_cp16(wle + gB + k0,      st + so + 24576);
    async_cp16(wle + gB + k0 + 16, st + so + 28672);
}
#else
// --- fallback: global -> VGPR -> ds_store ----------------------------------
struct StageRegs {
    v8bf ah0, ah1, al0, al1, bh0, bh1, bl0, bl1;
};

__device__ __forceinline__ void stage_fetch(StageRegs& r,
        const bf16_t* __restrict__ xh, const bf16_t* __restrict__ xl,
        const bf16_t* __restrict__ whe, const bf16_t* __restrict__ wle,
        int gA, int gB, int k0) {
    r.ah0 = *reinterpret_cast<const v8bf*>(xh  + gA + k0);
    r.ah1 = *reinterpret_cast<const v8bf*>(xh  + gA + k0 + 16);
    r.al0 = *reinterpret_cast<const v8bf*>(xl  + gA + k0);
    r.al1 = *reinterpret_cast<const v8bf*>(xl  + gA + k0 + 16);
    r.bh0 = *reinterpret_cast<const v8bf*>(whe + gB + k0);
    r.bh1 = *reinterpret_cast<const v8bf*>(whe + gB + k0 + 16);
    r.bl0 = *reinterpret_cast<const v8bf*>(wle + gB + k0);
    r.bl1 = *reinterpret_cast<const v8bf*>(wle + gB + k0 + 16);
}

__device__ __forceinline__ void stage_store(char* st, int so, const StageRegs& r) {
    *reinterpret_cast<v8bf*>(st + so)         = r.ah0;
    *reinterpret_cast<v8bf*>(st + so + 4096)  = r.ah1;
    *reinterpret_cast<v8bf*>(st + so + 8192)  = r.al0;
    *reinterpret_cast<v8bf*>(st + so + 12288) = r.al1;
    *reinterpret_cast<v8bf*>(st + so + 16384) = r.bh0;
    *reinterpret_cast<v8bf*>(st + so + 20480) = r.bh1;
    *reinterpret_cast<v8bf*>(st + so + 24576) = r.bl0;
    *reinterpret_cast<v8bf*>(st + so + 28672) = r.bl1;
}
#endif

__global__ void __launch_bounds__(256, 2)
moe_expert_gemm(const bf16_t* __restrict__ xh, const bf16_t* __restrict__ xl,
                const bf16_t* __restrict__ wh, const bf16_t* __restrict__ wl,
                const float* __restrict__ be,     // [E,O]
                const float* __restrict__ gate,   // [E,B]
                float* __restrict__ out) {        // [E,B,O]
    extern __shared__ char smem[];

    const int e    = blockIdx.z;
    const int tid  = threadIdx.x;
    const int wid  = tid >> 5;
    const int lane = tid & 31;
    const int half = lane >> 4;
    const int l15  = lane & 15;

    const int bm  = blockIdx.y * 128;
    const int bn  = blockIdx.x * 128;
    const int wmL = (wid >> 2) * 64;   // wave M origin within block
    const int wnL = (wid & 3) * 32;    // wave N origin within block

    const bf16_t* whe = wh + (size_t)e * OO * DD;
    const bf16_t* wle = wl + (size_t)e * OO * DD;

    // Staging mapping: thread -> (row = tid&127, chunk-plane cc = tid>>7);
    // each thread moves planes cc and cc+2 of its row for all 4 matrices.
    const int srow = tid & 127;
    const int scc  = tid >> 7;                    // 0|1
    const int gA   = (bm + srow) * DD + scc * 8;  // element offset into x
    const int gB   = (bn + srow) * DD + scc * 8;  // element offset into We[e]
    const int so   = scc * 2048 + srow * 16;      // byte offset in stage

    // Fragment read base offsets (bytes), loop-invariant.
    const int ra = half * 2048 + (wmL + l15) * 16;
    const int rb = 16384 + half * 2048 + (wnL + l15) * 16;

    char* st0 = smem;
    char* st1 = smem + STAGE_BYTES;

    v8f acc[4][2];
#pragma unroll
    for (int i = 0; i < 4; ++i)
#pragma unroll
        for (int j = 0; j < 2; ++j)
            acc[i][j] = (v8f){0.f,0.f,0.f,0.f,0.f,0.f,0.f,0.f};

#if defined(HAVE_ASYNC_LDS)
    stage_fetch_async(st0, so, xh, xl, whe, wle, gA, gB, 0);
    __builtin_amdgcn_s_wait_asynccnt(0);
    __syncthreads();

    for (int k0 = 0; k0 < DD; k0 += 64) {
        // substep A: DMA k0+32 -> stage1 while computing stage0 (k0)
        stage_fetch_async(st1, so, xh, xl, whe, wle, gA, gB, k0 + 32);
        stage_compute(st0, ra, rb, acc);
        __builtin_amdgcn_s_wait_asynccnt(0);
        __syncthreads();

        // substep B: DMA k0+64 -> stage0 while computing stage1 (k0+32)
        if (k0 + 64 < DD)
            stage_fetch_async(st0, so, xh, xl, whe, wle, gA, gB, k0 + 64);
        stage_compute(st1, ra, rb, acc);
        __builtin_amdgcn_s_wait_asynccnt(0);
        __syncthreads();
    }
#else
    StageRegs r;
    stage_fetch(r, xh, xl, whe, wle, gA, gB, 0);
    stage_store(st0, so, r);
    __syncthreads();

    for (int k0 = 0; k0 < DD; k0 += 64) {
        stage_fetch(r, xh, xl, whe, wle, gA, gB, k0 + 32);
        stage_compute(st0, ra, rb, acc);
        stage_store(st1, so, r);
        __syncthreads();

        if (k0 + 64 < DD) stage_fetch(r, xh, xl, whe, wle, gA, gB, k0 + 64);
        stage_compute(st1, ra, rb, acc);
        if (k0 + 64 < DD) stage_store(st0, so, r);
        __syncthreads();
    }
#endif

    // Epilogue: out[e,row,col] = gate[e,row] * relu(acc + be[e,col])
    const float* gate_e = gate + (size_t)e * BB;
    const float* be_e   = be + (size_t)e * OO;
#pragma unroll
    for (int j = 0; j < 2; ++j) {
        const int col  = bn + wnL + j * 16 + l15;
        const float bv = be_e[col];
#pragma unroll
        for (int i = 0; i < 4; ++i) {
            const int rbase = bm + wmL + i * 16 + half * 8;
#pragma unroll
            for (int q = 0; q < 8; ++q) {
                const int row = rbase + q;
                float v = acc[i][j][q] + bv;
                v = v > 0.f ? v : 0.f;
                v *= gate_e[row];
                __builtin_nontemporal_store(
                    v, out + ((size_t)e * BB + row) * OO + col);
            }
        }
    }
}

// ---------------------------------------------------------------------------
// Launch
// ---------------------------------------------------------------------------
extern "C" void kernel_launch(void* const* d_in, const int* in_sizes, int n_in,
                              void* d_out, int out_size, void* d_ws, size_t ws_size,
                              hipStream_t stream) {
    const float* x  = (const float*)d_in[0];   // [B,D]
    const float* Wr = (const float*)d_in[1];   // [E,D]
    const float* br = (const float*)d_in[2];   // [E]
    const float* We = (const float*)d_in[3];   // [E,O,D]
    const float* be = (const float*)d_in[4];   // [E,O]

    float* out_wo  = (float*)d_out;                        // [E,B,O]
    float* out_rl  = out_wo + (size_t)EE * BB * OO;        // [B,E]
    float* out_idx = out_rl + (size_t)BB * EE;             // [B,2] as float

    // Workspace layout (~84 MB total)
    bf16_t* x_hi  = (bf16_t*)d_ws;
    bf16_t* x_lo  = x_hi  + (size_t)BB * DD;
    bf16_t* we_hi = x_lo  + (size_t)BB * DD;
    bf16_t* we_lo = we_hi + (size_t)EE * OO * DD;
    float*  gate  = (float*)(we_lo + (size_t)EE * OO * DD); // [E,B]

    moe_split_bf16<<<2048, 256, 0, stream>>>(x,  x_hi,  x_lo,  BB * DD);
    moe_split_bf16<<<4096, 256, 0, stream>>>(We, we_hi, we_lo, EE * OO * DD);

    moe_router<<<BB / 256, 256, 0, stream>>>(x, Wr, br, out_rl, out_idx, gate);

    dim3 grid(OO / 128, BB / 128, EE);
    moe_expert_gemm<<<grid, 256, 2 * STAGE_BYTES, stream>>>(
        x_hi, x_lo, we_hi, we_lo, be, gate, out_wo);
}
